// TemporalAttention_386547056826
// MI455X (gfx1250) — matrix operand order repaired and checked by reading, not compiled, not run on hardware
//
#include <hip/hip_runtime.h>
#include <hip/hip_bf16.h>
#include <math.h>

// Problem constants (reference: B=2, T=2048, DIM=1024, H=16, HD=64)
#define BB   2
#define TT   2048
#define DIMD 1024
#define HH   16
#define HDD  64

typedef __attribute__((ext_vector_type(16))) _Float16 v16h;
typedef __attribute__((ext_vector_type(8)))  _Float16 v8h;
typedef __attribute__((ext_vector_type(4)))  _Float16 v4h;
typedef __attribute__((ext_vector_type(2)))  _Float16 v2h;
typedef __attribute__((ext_vector_type(8)))  float    v8f;
typedef __attribute__((ext_vector_type(4)))  int      v4i;

// ---- Async global->LDS copy (gfx1250), with graceful fallback --------------
#if defined(__has_builtin)
#if __has_builtin(__builtin_amdgcn_global_load_async_to_lds_b128)
#define USE_ASYNC_COPY 1
#endif
#endif

#if defined(USE_ASYNC_COPY)
// ROCm 7.2 signature (from probe diagnostic): param 1 is a generic `int4*`.
// LDS dst passed as AS(3) int4* (implicitly converts to AS3 or generic param).
typedef __attribute__((address_space(3))) v4i lds_v4i;
__device__ __forceinline__ void async_cp16(const void* g, void* l) {
  __builtin_amdgcn_global_load_async_to_lds_b128(
      (v4i*)const_cast<void*>(g), (lds_v4i*)l, 0, 0);
}
__device__ __forceinline__ void async_wait0() {
#if __has_builtin(__builtin_amdgcn_s_wait_asynccnt)
  __builtin_amdgcn_s_wait_asynccnt(0);
#else
  asm volatile("s_wait_asynccnt 0x0" ::: "memory");
#endif
}
#else
__device__ __forceinline__ void async_cp16(const void* g, void* l) {
  *reinterpret_cast<v8h*>(l) = *reinterpret_cast<const v8h*>(g);
}
__device__ __forceinline__ void async_wait0() {}
#endif

__device__ __forceinline__ v8f wmma_f16(v16h a, v16h b, v8f c) {
  return __builtin_amdgcn_wmma_f32_16x16x32_f16(
      false, a, false, b, (short)0, c, false, false);
}

// ---- Fragment gathers (CDNA5 16x16x32 f16 VGPR layouts) --------------------
// A (16x32 row-major, stride in halves). lane L: row=L&15,
// element e: K = (e>=8?16:0) + (L&16?8:0) + (e&7)  -> two 16B runs per lane.
__device__ __forceinline__ v16h frag_a(const _Float16* base, int stride, int lane) {
  v16h a;
  const _Float16* p = base + (lane & 15) * stride + ((lane & 16) ? 8 : 0);
#pragma unroll
  for (int e = 0; e < 16; ++e)
    a[e] = p[((e >> 3) << 4) + (e & 7)];
  return a;
}
// B = X^T where X row-major [n][k] (B[k][n] = X[n][k]). lane L: col=L&15,
// element e: K = (e>=8?16:0) + (L&16?8:0) + (e&7)  -> two 16B runs per lane.
__device__ __forceinline__ v16h frag_bt(const _Float16* base, int stride, int lane) {
  v16h b;
  const _Float16* p = base + (lane & 15) * stride + ((lane & 16) ? 8 : 0);
#pragma unroll
  for (int e = 0; e < 16; ++e)
    b[e] = p[((e >> 3) << 4) + (e & 7)];
  return b;
}

__device__ __forceinline__ v8h cvt8(float4 f0, float4 f1) {
  v8h h;
  h[0]=(_Float16)f0.x; h[1]=(_Float16)f0.y; h[2]=(_Float16)f0.z; h[3]=(_Float16)f0.w;
  h[4]=(_Float16)f1.x; h[5]=(_Float16)f1.y; h[6]=(_Float16)f1.z; h[7]=(_Float16)f1.w;
  return h;
}

// Stage a 32(K) x 128(N) fp32 weight tile as f16 transposed [n][k] in LDS.
// Each thread: 4x4 patch via 4 coalesced float4 loads + 4 ds_store_b64.
__device__ __forceinline__ void stage_w_t(const float* __restrict__ W, int ldw,
                                          int k0, int n0, int tid,
                                          _Float16 (*Bs)[40]) {
  const int kq = (tid >> 5) * 4;   // 0..28
  const int nq = (tid & 31) * 4;   // 0..124
  const float* wb = &W[(size_t)(k0 + kq) * ldw + n0 + nq];
  float4 r0 = *reinterpret_cast<const float4*>(wb);
  float4 r1 = *reinterpret_cast<const float4*>(wb + ldw);
  float4 r2 = *reinterpret_cast<const float4*>(wb + 2 * (size_t)ldw);
  float4 r3 = *reinterpret_cast<const float4*>(wb + 3 * (size_t)ldw);
  v4h c;
  c[0]=(_Float16)r0.x; c[1]=(_Float16)r1.x; c[2]=(_Float16)r2.x; c[3]=(_Float16)r3.x;
  *reinterpret_cast<v4h*>(&Bs[nq][kq]) = c;
  c[0]=(_Float16)r0.y; c[1]=(_Float16)r1.y; c[2]=(_Float16)r2.y; c[3]=(_Float16)r3.y;
  *reinterpret_cast<v4h*>(&Bs[nq + 1][kq]) = c;
  c[0]=(_Float16)r0.z; c[1]=(_Float16)r1.z; c[2]=(_Float16)r2.z; c[3]=(_Float16)r3.z;
  *reinterpret_cast<v4h*>(&Bs[nq + 2][kq]) = c;
  c[0]=(_Float16)r0.w; c[1]=(_Float16)r1.w; c[2]=(_Float16)r2.w; c[3]=(_Float16)r3.w;
  *reinterpret_cast<v4h*>(&Bs[nq + 3][kq]) = c;
}

// ============================================================================
// Kernel 1: QKV GEMM (4096x3072x1024) + RoPE epilogue.
// Block 256 thr (8 waves); block tile 64x128; wave tile 32x32 (2x2 WMMA).
// ============================================================================
__global__ __launch_bounds__(256) void qkv_rope_kernel(
    const float* __restrict__ x, const float* __restrict__ Wqkv,
    _Float16* __restrict__ Qo, _Float16* __restrict__ Ko, _Float16* __restrict__ Vo)
{
  __shared__ _Float16 As[64][40];    // [m][k]
  __shared__ _Float16 Bs[128][40];   // transposed [n][k]
  __shared__ float    Cs[64][132];   // block C tile (for RoPE pairing)

  const int tid  = threadIdx.x;
  const int lane = tid & 31;
  const int w    = tid >> 5;
  const int wm   = w & 1, wn = w >> 1;      // wave grid 2(M) x 4(N)
  const int m0   = blockIdx.y * 64;
  const int n0   = blockIdx.x * 128;

  const int am = tid >> 2;            // 0..63
  const int ak = (tid & 3) * 8;       // 0,8,16,24

  v8f acc00 = {}, acc01 = {}, acc10 = {}, acc11 = {};

  for (int k0 = 0; k0 < DIMD; k0 += 32) {
    { // A: 64x32 fp32 -> f16
      const float4* p = reinterpret_cast<const float4*>(
          &x[(size_t)(m0 + am) * DIMD + k0 + ak]);
      *reinterpret_cast<v8h*>(&As[am][ak]) = cvt8(p[0], p[1]);
    }
    stage_w_t(Wqkv, 3 * DIMD, k0, n0, tid, Bs);
    __syncthreads();
    v16h a0 = frag_a (&As[wm * 32][0],      40, lane);
    v16h a1 = frag_a (&As[wm * 32 + 16][0], 40, lane);
    v16h b0 = frag_bt(&Bs[wn * 32][0],      40, lane);
    v16h b1 = frag_bt(&Bs[wn * 32 + 16][0], 40, lane);
    acc00 = wmma_f16(a0, b0, acc00);
    acc01 = wmma_f16(a0, b1, acc01);
    acc10 = wmma_f16(a1, b0, acc10);
    acc11 = wmma_f16(a1, b1, acc11);
    __syncthreads();
  }

  // Stage full block C tile so epilogue can access RoPE pair partners.
#pragma unroll
  for (int j = 0; j < 8; ++j) {
    int r = j + ((lane & 16) ? 8 : 0);
    int c = lane & 15;
    Cs[wm * 32 + r     ][wn * 32 + c     ] = acc00[j];
    Cs[wm * 32 + r     ][wn * 32 + 16 + c] = acc01[j];
    Cs[wm * 32 + 16 + r][wn * 32 + c     ] = acc10[j];
    Cs[wm * 32 + 16 + r][wn * 32 + 16 + c] = acc11[j];
  }
  __syncthreads();

  // Elementwise epilogue over 64x128: process (even,odd) column pairs.
  const float kln = -0.2878231366242558f;  // -ln(10000)/32
#pragma unroll
  for (int i = 0; i < 16; ++i) {
    int idx = tid + 256 * i;        // 0..4095 pairs
    int r   = idx >> 6;             // 0..63
    int cp  = (idx & 63) * 2;       // even col 0..126
    int gr  = m0 + r;
    int gc  = n0 + cp;
    int bi  = gr >> 11;
    int t   = gr & (TT - 1);
    int sec = gc >> 10;             // 0=q 1=k 2=v
    int d   = gc & (DIMD - 1);
    int h   = d >> 6;
    int hd  = d & (HDD - 1);        // even
    float ve = Cs[r][cp], vo = Cs[r][cp + 1];
    size_t oidx = (((size_t)(bi * HH + h) * TT) + t) * HDD + hd;
    if (sec == 2) {
      v2h pk; pk[0] = (_Float16)ve; pk[1] = (_Float16)vo;
      *reinterpret_cast<v2h*>(&Vo[oidx]) = pk;
    } else {
      // rotate_half interleaved: rot[2i] = -x[2i+1], rot[2i+1] = x[2i]
      float fe = __expf(kln * (float)(hd & 31));
      float fo = __expf(kln * (float)((hd + 1) & 31));
      float ae = (float)t * fe, ao = (float)t * fo;
      float oe = ve * __cosf(ae) - vo * __sinf(ae);
      float oo = vo * __cosf(ao) + ve * __sinf(ao);
      v2h pk;
      if (sec == 0) {
        pk[0] = (_Float16)(oe * 0.125f);   // pre-apply HD^-0.5
        pk[1] = (_Float16)(oo * 0.125f);
        *reinterpret_cast<v2h*>(&Qo[oidx]) = pk;
      } else {
        pk[0] = (_Float16)oe; pk[1] = (_Float16)oo;
        *reinterpret_cast<v2h*>(&Ko[oidx]) = pk;
      }
    }
  }
}

// ============================================================================
// Kernel 2: Flash attention per (b,h). 64 Q rows/block, key blocks of 64.
// Q/K staged via async global->LDS copies; V staged transposed for contiguous
// B-fragments; 8 WMMAs per wave per key block; parallel softmax stats.
// ============================================================================
__global__ __launch_bounds__(256) void attn_kernel(
    const _Float16* __restrict__ Q, const _Float16* __restrict__ Kb,
    const _Float16* __restrict__ V, _Float16* __restrict__ O)
{
  __shared__ _Float16 qs [64][72];
  __shared__ _Float16 ks [64][72];
  __shared__ _Float16 vts[64][72];   // transposed: [hd][key]
  __shared__ float    ss [64][68];
  __shared__ _Float16 ps [64][72];
  __shared__ float    mrow[64], lrow[64], arow[64];
  __shared__ float    pmax[64][4], psum[64][4];

  const int tid  = threadIdx.x;
  const int lane = tid & 31;
  const int w    = tid >> 5;
  const int wr   = w & 3;
  const int wc   = w >> 2;
  const int q0   = blockIdx.x * 64;
  const size_t base = (size_t)blockIdx.y * TT * HDD;

  const int sr = tid >> 2;          // 0..63
  const int sc = (tid & 3) * 16;    // 0,16,32,48

  // Q block: async copy straight into LDS (no VGPR round trip).
  async_cp16(&Q[base + (size_t)(q0 + sr) * HDD + sc],     &qs[sr][sc]);
  async_cp16(&Q[base + (size_t)(q0 + sr) * HDD + sc + 8], &qs[sr][sc + 8]);
  if (tid < 64) { mrow[tid] = -1e30f; lrow[tid] = 0.f; }

  v8f o0 = {}, o1 = {};

#pragma unroll 1
  for (int kb = 0; kb < TT; kb += 64) {
    __syncthreads();
    // K rows: async copy. V: manual transpose into [hd][key].
    async_cp16(&Kb[base + (size_t)(kb + sr) * HDD + sc],     &ks[sr][sc]);
    async_cp16(&Kb[base + (size_t)(kb + sr) * HDD + sc + 8], &ks[sr][sc + 8]);
    {
      const v8h* vp = reinterpret_cast<const v8h*>(&V[base + (size_t)(kb + sr) * HDD + sc]);
      v8h v0 = vp[0], v1 = vp[1];
#pragma unroll
      for (int i = 0; i < 8; ++i) {
        vts[sc + i][sr]     = v0[i];
        vts[sc + 8 + i][sr] = v1[i];
      }
    }
    if (kb + 64 < TT) {   // prefetch next K/V block (global_prefetch_b8)
      __builtin_prefetch(&Kb[base + (size_t)(kb + 64 + sr) * HDD + sc], 0, 1);
      __builtin_prefetch(&V [base + (size_t)(kb + 64 + sr) * HDD + sc], 0, 1);
    }
    async_wait0();
    __syncthreads();

    // S = Q K^T : wave computes rows wr*16.., keys wc*32..(+16); K-dim = 64
    v8f s0 = {}, s1 = {};
#pragma unroll
    for (int kk = 0; kk < HDD; kk += 32) {
      v16h a  = frag_a (&qs[wr * 16][kk],      72, lane);
      v16h b0 = frag_bt(&ks[wc * 32][kk],      72, lane);
      v16h b1 = frag_bt(&ks[wc * 32 + 16][kk], 72, lane);
      s0 = wmma_f16(a, b0, s0);
      s1 = wmma_f16(a, b1, s1);
    }
#pragma unroll
    for (int j = 0; j < 8; ++j) {
      int r = wr * 16 + j + ((lane & 16) ? 8 : 0);
      ss[r][wc * 32 + (lane & 15)]      = s0[j];
      ss[r][wc * 32 + 16 + (lane & 15)] = s1[j];
    }
    __syncthreads();

    // stats phase 1: partial max (4 threads per row, 16 cols each)
    {
      float m = ss[sr][sc];
#pragma unroll
      for (int c = 1; c < 16; ++c) m = fmaxf(m, ss[sr][sc + c]);
      pmax[sr][tid & 3] = m;
    }
    __syncthreads();
    // phase 2: new max, exponentials, partial sums
    {
      float mn = fmaxf(fmaxf(pmax[sr][0], pmax[sr][1]),
                       fmaxf(pmax[sr][2], pmax[sr][3]));
      mn = fmaxf(mn, mrow[sr]);
      float sum = 0.f;
#pragma unroll
      for (int c = 0; c < 16; ++c) {
        float p = __expf(ss[sr][sc + c] - mn);
        ps[sr][sc + c] = (_Float16)p;
        sum += p;
      }
      psum[sr][tid & 3] = sum;
      if ((tid & 3) == 0) {
        float al = __expf(mrow[sr] - mn);
        arow[sr] = al;
        mrow[sr] = mn;
      }
    }
    __syncthreads();
    if ((tid & 3) == 0)
      lrow[sr] = lrow[sr] * arow[sr]
               + psum[sr][0] + psum[sr][1] + psum[sr][2] + psum[sr][3];

    // rescale running O, then O += P @ V
#pragma unroll
    for (int j = 0; j < 8; ++j) {
      int r = wr * 16 + j + ((lane & 16) ? 8 : 0);
      float al = arow[r];
      o0[j] *= al;
      o1[j] *= al;
    }
#pragma unroll
    for (int kk = 0; kk < 64; kk += 32) {
      v16h a  = frag_a (&ps[wr * 16][kk],       72, lane);
      v16h b0 = frag_bt(&vts[wc * 32][kk],      72, lane);
      v16h b1 = frag_bt(&vts[wc * 32 + 16][kk], 72, lane);
      o0 = wmma_f16(a, b0, o0);
      o1 = wmma_f16(a, b1, o1);
    }
  }
  __syncthreads();

  const int b = blockIdx.y >> 4, h = blockIdx.y & (HH - 1);
#pragma unroll
  for (int j = 0; j < 8; ++j) {
    int r = wr * 16 + j + ((lane & 16) ? 8 : 0);
    float inv = 1.f / lrow[r];
    int t = q0 + r;
    size_t rowb = ((size_t)b * TT + t) * DIMD + h * HDD + wc * 32 + (lane & 15);
    O[rowb]      = (_Float16)(o0[j] * inv);
    O[rowb + 16] = (_Float16)(o1[j] * inv);
  }
}

// ============================================================================
// Kernel 3: output projection GEMM (4096x1024x1024) + bias, f32 out.
// ============================================================================
__global__ __launch_bounds__(256) void out_proj_kernel(
    const _Float16* __restrict__ A, const float* __restrict__ Wout,
    const float* __restrict__ bout, float* __restrict__ out)
{
  __shared__ _Float16 As[64][40];
  __shared__ _Float16 Bs[128][40];

  const int tid  = threadIdx.x;
  const int lane = tid & 31;
  const int w    = tid >> 5;
  const int wm   = w & 1, wn = w >> 1;
  const int m0   = blockIdx.y * 64;
  const int n0   = blockIdx.x * 128;

  const int am = tid >> 2;
  const int ak = (tid & 3) * 8;

  v8f acc00 = {}, acc01 = {}, acc10 = {}, acc11 = {};

  for (int k0 = 0; k0 < DIMD; k0 += 32) {
    // A tile is already f16: async copy straight to LDS.
    async_cp16(&A[(size_t)(m0 + am) * DIMD + k0 + ak], &As[am][ak]);
    stage_w_t(Wout, DIMD, k0, n0, tid, Bs);
    async_wait0();
    __syncthreads();
    v16h a0 = frag_a (&As[wm * 32][0],      40, lane);
    v16h a1 = frag_a (&As[wm * 32 + 16][0], 40, lane);
    v16h b0 = frag_bt(&Bs[wn * 32][0],      40, lane);
    v16h b1 = frag_bt(&Bs[wn * 32 + 16][0], 40, lane);
    acc00 = wmma_f16(a0, b0, acc00);
    acc01 = wmma_f16(a0, b1, acc01);
    acc10 = wmma_f16(a1, b0, acc10);
    acc11 = wmma_f16(a1, b1, acc11);
    __syncthreads();
  }

  const int c0 = n0 + wn * 32 + (lane & 15);
  const float bc0 = bout[c0], bc1 = bout[c0 + 16];
#pragma unroll
  for (int j = 0; j < 8; ++j) {
    int r = m0 + wm * 32 + j + ((lane & 16) ? 8 : 0);
    out[(size_t)r * DIMD + c0]             = acc00[j] + bc0;
    out[(size_t)r * DIMD + c0 + 16]        = acc01[j] + bc1;
    out[(size_t)(r + 16) * DIMD + c0]      = acc10[j] + bc0;
    out[(size_t)(r + 16) * DIMD + c0 + 16] = acc11[j] + bc1;
  }
}

// ============================================================================
extern "C" void kernel_launch(void* const* d_in, const int* in_sizes, int n_in,
                              void* d_out, int out_size, void* d_ws, size_t ws_size,
                              hipStream_t stream) {
  const float* x    = (const float*)d_in[0];
  const float* Wqkv = (const float*)d_in[1];
  const float* Wout = (const float*)d_in[2];
  const float* bout = (const float*)d_in[3];
  float* out = (float*)d_out;

  const size_t per = (size_t)BB * HH * TT * HDD;   // 4,194,304 f16 elements
  _Float16* Q  = (_Float16*)d_ws;
  _Float16* K  = Q + per;
  _Float16* V  = K + per;
  _Float16* AO = V + per;

  qkv_rope_kernel<<<dim3((3 * DIMD) / 128, (BB * TT) / 64), 256, 0, stream>>>(
      x, Wqkv, Q, K, V);
  attn_kernel<<<dim3(TT / 64, BB * HH), 256, 0, stream>>>(Q, K, V, AO);
  out_proj_kernel<<<dim3(DIMD / 128, (BB * TT) / 64), 256, 0, stream>>>(
      AO, Wout, bout, out);
}
